// SCUNet_aaf6aa_85066122264645
// MI455X (gfx1250) — matrix operand order repaired and checked by reading, not compiled, not run on hardware
//
#include <hip/hip_runtime.h>

// ---------------------------------------------------------------------------
// CDNA5 (gfx1250) fused Swin shifted-window attention.
// wave32; WMMA bf16 16x16x32; one workgroup per 8x8 window (64 tok x 256 ch).
// ---------------------------------------------------------------------------

typedef __attribute__((ext_vector_type(16))) __bf16 v16bf;
typedef __attribute__((ext_vector_type(8)))  float  v8f;
typedef __attribute__((ext_vector_type(2)))  float  v2f;
typedef __attribute__((ext_vector_type(2)))  __bf16 v2bf;

union AU { unsigned int u[8]; v16bf v; };

// Native RNE converts: backend emits v_cvt_*bf16_f32 (packed where possible).
__device__ __forceinline__ unsigned short f2bf(float f) {
    __bf16 h = (__bf16)f;
    return __builtin_bit_cast(unsigned short, h);
}
__device__ __forceinline__ unsigned int pack2bf(float a, float b) {
    v2f t = {a, b};
    v2bf c = __builtin_convertvector(t, v2bf);
    return __builtin_bit_cast(unsigned int, c);
}

// LDS layout (dynamic):
//   [0, 100352)           : QKV bf16, 64 rows x 784 (768 + 16 pad) u16
//   [100352, 100352+40960): union { Xstage 64x272 bf16 | PStage 8x64x40 bf16
//                                   | AttnOut 64x272 bf16 }
#define QKV_STRIDE 784
#define X_STRIDE   272
#define P_STRIDE   40
#define UNION_OFF  100352
#define SMEM_BYTES (100352 + 40960)

__device__ __forceinline__ v8f wmma_bf16(const AU& a, const AU& b, v8f c) {
    return __builtin_amdgcn_wmma_f32_16x16x32_bf16(
        false, a.v, false, b.v, (short)0, c, false, false);
}

// ---------------------------------------------------------------------------
// Prep kernel 1: fp32 weights -> bf16, K-pair interleaved:
//   Wb[(k>>1)*2N + 2n + (k&1)] = bf16(W[k*N + n])
// so one u32 load gives the (k, k+1)@n pair a WMMA-B lane wants.
// ---------------------------------------------------------------------------
__global__ void swin_pack_w(const float* __restrict__ W,
                            unsigned short* __restrict__ Wb, int N, int total) {
    int idx = blockIdx.x * blockDim.x + threadIdx.x;
    if (idx >= total) return;
    int k = idx / N, n = idx - k * N;
    Wb[(k >> 1) * (2 * N) + 2 * n + (k & 1)] = f2bf(W[idx]);
}

// ---------------------------------------------------------------------------
// Prep kernel 2: bias[h][p][q] = rel_pos[h, py-qy+7, px-qx+7]  (8x64x64 f32)
// ---------------------------------------------------------------------------
__global__ void swin_bias(const float* __restrict__ rel_pos,
                          float* __restrict__ bias) {
    int idx = blockIdx.x * blockDim.x + threadIdx.x;   // 32768 total
    int h = idx >> 12, p = (idx >> 6) & 63, q = idx & 63;
    int ri = (p >> 3) - (q >> 3) + 7;
    int rj = (p & 7) - (q & 7) + 7;
    bias[idx] = rel_pos[h * 225 + ri * 15 + rj];
}

// ---------------------------------------------------------------------------
// Main fused kernel: grid = 4*1024 windows, 256 threads (8 waves).
// ---------------------------------------------------------------------------
__global__ __launch_bounds__(256)
void swin_window_kernel(const float* __restrict__ x,
                        const unsigned short* __restrict__ Wqkv_bf,
                        const float* __restrict__ b_qkv,
                        const unsigned short* __restrict__ Wout_bf,
                        const float* __restrict__ b_out,
                        const float* __restrict__ bias,
                        float* __restrict__ out) {
    extern __shared__ char smem[];
    unsigned short* QKVs = (unsigned short*)smem;
    unsigned short* XsU  = (unsigned short*)(smem + UNION_OFF);

    const int blk  = blockIdx.x;
    const int bi   = blk >> 10;
    const int w    = blk & 1023;
    const int wy   = w >> 5, wx = w & 31;
    const int tid  = threadIdx.x;
    const int lane = tid & 31;
    const int wave = tid >> 5;
    const int hh   = lane >> 4;     // lane half (0/1)
    const int ln   = lane & 15;

    // Warm L2 for the packed weights (global_prefetch_b8 path).
    __builtin_prefetch(Wqkv_bf + ((size_t)tid << 8), 0, 3);
    __builtin_prefetch(Wout_bf + ((size_t)tid << 7), 0, 3);

    const float* xb = x + (size_t)bi * (256u * 256u * 256u);

    // ---------------- Phase 0: gather rolled window -> LDS bf16 -------------
    for (int i = 0; i < 16; ++i) {
        int linear = tid + i * 256;            // 4096 float4 slots
        int tok = linear >> 6;                 // 64 float4 per token
        int c4  = linear & 63;
        int py = tok >> 3, px = tok & 7;
        int gy = (wy * 8 + py + 4) & 255;
        int gx = (wx * 8 + px + 4) & 255;
        const float4 v4 = *(const float4*)(xb + ((size_t)gy * 256 + gx) * 256 + c4 * 4);
        unsigned int* dst = (unsigned int*)(XsU + tok * X_STRIDE + c4 * 4);
        dst[0] = pack2bf(v4.x, v4.y);
        dst[1] = pack2bf(v4.z, v4.w);
    }
    __syncthreads();

    // ---------------- Phase 1: QKV = X(64x256) @ Wqkv(256x768) + b ----------
    // n-tiles in chunks of 3 so A-tiles are reused across the n-loop.
    {
        const unsigned int* X32 = (const unsigned int*)XsU;
        const unsigned int* W32 = (const unsigned int*)Wqkv_bf;
        for (int chunk = 0; chunk < 2; ++chunk) {
            int ntBase = wave * 6 + chunk * 3;      // 0..47 in steps of 3
            v8f acc[3][4] = {};
            for (int k = 0; k < 8; ++k) {
                AU A[4];
#pragma unroll
                for (int mt = 0; mt < 4; ++mt) {
                    int row32 = ((mt * 16 + ln) * X_STRIDE) >> 1;
#pragma unroll
                    for (int j = 0; j < 8; ++j) {
                        int kk = hh * 16 + 2 * j;
                        A[mt].u[j] = X32[row32 + ((k * 32 + kk) >> 1)];
                    }
                }
#pragma unroll
                for (int t = 0; t < 3; ++t) {
                    int nIdx = (ntBase + t) * 16 + ln;
                    AU B;
#pragma unroll
                    for (int j = 0; j < 8; ++j) {
                        int pairIdx = k * 16 + hh * 8 + j;
                        B.u[j] = W32[pairIdx * 768 + nIdx];
                    }
#pragma unroll
                    for (int mt = 0; mt < 4; ++mt)
                        acc[t][mt] = wmma_bf16(A[mt], B, acc[t][mt]);
                }
            }
#pragma unroll
            for (int t = 0; t < 3; ++t) {
                int nIdx = (ntBase + t) * 16 + ln;
                float bq = b_qkv[nIdx];
#pragma unroll
                for (int mt = 0; mt < 4; ++mt)
#pragma unroll
                    for (int r = 0; r < 8; ++r) {
                        int m = mt * 16 + r + 8 * hh;
                        QKVs[m * QKV_STRIDE + nIdx] = f2bf(acc[t][mt][r] + bq);
                    }
            }
        }
    }
    __syncthreads();

    // ---------------- Phase 2: attention, wave == head ----------------------
    const int h = wave;
    const float scale = 0.17677669529663687f;   // 1/sqrt(32)
    const unsigned int* Q32 = (const unsigned int*)QKVs;

    // A-tiles of q (16x32 bf16 each), cached in regs
    AU Aq[4];
#pragma unroll
    for (int mt = 0; mt < 4; ++mt) {
        int row32 = (mt * 16 + ln) * (QKV_STRIDE / 2);
#pragma unroll
        for (int j = 0; j < 8; ++j) {
            int kk = hh * 16 + 2 * j;
            Aq[mt].u[j] = Q32[row32 + ((h * 32 + kk) >> 1)];
        }
    }

    // sim = q @ k^T : B[k][n] = K[n][k] -> contiguous row-major reads of k
    v8f sim[4][4];
#pragma unroll
    for (int nt = 0; nt < 4; ++nt) {
        AU B;
        int row32 = (nt * 16 + ln) * (QKV_STRIDE / 2);
#pragma unroll
        for (int j = 0; j < 8; ++j) {
            int kk = hh * 16 + 2 * j;
            B.u[j] = Q32[row32 + ((256 + h * 32 + kk) >> 1)];
        }
#pragma unroll
        for (int mt = 0; mt < 4; ++mt) {
            v8f z = {};
            sim[mt][nt] = wmma_bf16(Aq[mt], B, z);
        }
    }

    // scale + rel-pos bias + shift mask
    const bool maskY = (wy == 31), maskX = (wx == 31);
    const float* biasH = bias + h * 4096;
#pragma unroll
    for (int mt = 0; mt < 4; ++mt)
#pragma unroll
        for (int nt = 0; nt < 4; ++nt)
#pragma unroll
            for (int r = 0; r < 8; ++r) {
                int p = mt * 16 + r + 8 * hh;
                int q = nt * 16 + ln;
                float v = sim[mt][nt][r] * scale + biasH[p * 64 + q];
                int py = p >> 3, px = p & 7, qy = q >> 3, qx = q & 7;
                bool msk = (maskY && ((py < 4) != (qy < 4))) ||
                           (maskX && ((px < 4) != (qx < 4)));
                sim[mt][nt][r] = msk ? -1e30f : v;
            }

    // row softmax: row lives in one 16-lane half -> xor-shuffle widths 1..8
#pragma unroll
    for (int mt = 0; mt < 4; ++mt)
#pragma unroll
        for (int r = 0; r < 8; ++r) {
            float mx = -1e30f;
#pragma unroll
            for (int nt = 0; nt < 4; ++nt) mx = fmaxf(mx, sim[mt][nt][r]);
#pragma unroll
            for (int o = 1; o < 16; o <<= 1) mx = fmaxf(mx, __shfl_xor(mx, o, 32));
            float sum = 0.f;
#pragma unroll
            for (int nt = 0; nt < 4; ++nt) {
                float e = __expf(sim[mt][nt][r] - mx);
                sim[mt][nt][r] = e;
                sum += e;
            }
#pragma unroll
            for (int o = 1; o < 16; o <<= 1) sum += __shfl_xor(sum, o, 32);
            float inv = 1.0f / sum;
#pragma unroll
            for (int nt = 0; nt < 4; ++nt) sim[mt][nt][r] *= inv;
        }

    // out = probs @ v  (K=64 in two 32-chunks; probs staged C-layout->A-layout)
    unsigned short* myPS = XsU + h * 64 * P_STRIDE;   // per-wave stage
    v8f oacc[4][2] = {};
    for (int kk = 0; kk < 2; ++kk) {
#pragma unroll
        for (int mt = 0; mt < 4; ++mt)
#pragma unroll
            for (int nt = kk * 2; nt < kk * 2 + 2; ++nt)
#pragma unroll
                for (int r = 0; r < 8; ++r) {
                    int p = mt * 16 + r + 8 * hh;
                    int qloc = nt * 16 + ln - kk * 32;
                    myPS[p * P_STRIDE + qloc] = f2bf(sim[mt][nt][r]);
                }
        asm volatile("s_wait_dscnt 0x0" ::: "memory");   // per-wave LDS RAW

        const unsigned int* P32 = (const unsigned int*)myPS;
#pragma unroll
        for (int nt2 = 0; nt2 < 2; ++nt2) {
            AU B;                                        // B[k][n] = v[tok][dim]
            int ch = 512 + h * 32 + nt2 * 16 + ln;
#pragma unroll
            for (int j = 0; j < 8; ++j) {
                int t0 = kk * 32 + hh * 16 + 2 * j;
                unsigned int lo = QKVs[t0 * QKV_STRIDE + ch];
                unsigned int hi = QKVs[(t0 + 1) * QKV_STRIDE + ch];
                B.u[j] = lo | (hi << 16);
            }
#pragma unroll
            for (int mt = 0; mt < 4; ++mt) {
                AU A;
                int rowb = (mt * 16 + ln) * P_STRIDE;
#pragma unroll
                for (int j = 0; j < 8; ++j) {
                    int kx = hh * 16 + 2 * j;
                    A.u[j] = P32[(rowb + kx) >> 1];
                }
                oacc[mt][nt2] = wmma_bf16(A, B, oacc[mt][nt2]);
            }
        }
    }
    __syncthreads();                       // everyone done with PStage

    // write attention output (bf16) into union region, stride 272
    unsigned short* AO = XsU;
#pragma unroll
    for (int mt = 0; mt < 4; ++mt)
#pragma unroll
        for (int nt2 = 0; nt2 < 2; ++nt2)
#pragma unroll
            for (int r = 0; r < 8; ++r) {
                int m = mt * 16 + r + 8 * hh;
                AO[m * X_STRIDE + h * 32 + nt2 * 16 + ln] = f2bf(oacc[mt][nt2][r]);
            }
    __syncthreads();

    // ---------------- Phase 3: Y = AO(64x256) @ Wout(256x256) + b -----------
    {
        const unsigned int* A32  = (const unsigned int*)AO;
        const unsigned int* WO32 = (const unsigned int*)Wout_bf;
        float* ob = out + (size_t)bi * (256u * 256u * 256u);
        v8f acc[2][4] = {};
        for (int k = 0; k < 8; ++k) {
            AU A[4];
#pragma unroll
            for (int mt = 0; mt < 4; ++mt) {
                int row32 = ((mt * 16 + ln) * X_STRIDE) >> 1;
#pragma unroll
                for (int j = 0; j < 8; ++j) {
                    int kx = hh * 16 + 2 * j;
                    A[mt].u[j] = A32[row32 + ((k * 32 + kx) >> 1)];
                }
            }
#pragma unroll
            for (int nt2 = 0; nt2 < 2; ++nt2) {
                int n = (wave * 2 + nt2) * 16 + ln;
                AU B;
#pragma unroll
                for (int j = 0; j < 8; ++j) {
                    int pairIdx = k * 16 + hh * 8 + j;
                    B.u[j] = WO32[pairIdx * 256 + n];
                }
#pragma unroll
                for (int mt = 0; mt < 4; ++mt)
                    acc[nt2][mt] = wmma_bf16(A[mt], B, acc[nt2][mt]);
            }
        }
#pragma unroll
        for (int nt2 = 0; nt2 < 2; ++nt2) {
            int n = (wave * 2 + nt2) * 16 + ln;
            float bo = b_out[n];
#pragma unroll
            for (int mt = 0; mt < 4; ++mt)
#pragma unroll
                for (int r = 0; r < 8; ++r) {
                    int p = mt * 16 + r + 8 * hh;
                    int gy = (wy * 8 + (p >> 3) + 4) & 255;
                    int gx = (wx * 8 + (p & 7) + 4) & 255;
                    ob[((size_t)gy * 256 + gx) * 256 + n] = acc[nt2][mt][r] + bo;
                }
        }
    }
}

// ---------------------------------------------------------------------------
extern "C" void kernel_launch(void* const* d_in, const int* in_sizes, int n_in,
                              void* d_out, int out_size, void* d_ws, size_t ws_size,
                              hipStream_t stream) {
    const float* x       = (const float*)d_in[0];
    const float* W_qkv   = (const float*)d_in[1];
    const float* b_qkv   = (const float*)d_in[2];
    const float* rel_pos = (const float*)d_in[3];
    const float* W_out   = (const float*)d_in[4];
    const float* b_out   = (const float*)d_in[5];
    float* out = (float*)d_out;

    char* ws = (char*)d_ws;
    unsigned short* Wqkv_bf = (unsigned short*)ws;                      // 393,216 B
    unsigned short* Wout_bf = (unsigned short*)(ws + 393216);           // 131,072 B
    float*          biasBuf = (float*)(ws + 393216 + 131072);           // 131,072 B

    swin_pack_w<<<(256 * 768 + 255) / 256, 256, 0, stream>>>(W_qkv, Wqkv_bf, 768, 256 * 768);
    swin_pack_w<<<(256 * 256 + 255) / 256, 256, 0, stream>>>(W_out, Wout_bf, 256, 256 * 256);
    swin_bias  <<<(8 * 64 * 64) / 256, 256, 0, stream>>>(rel_pos, biasBuf);

    swin_window_kernel<<<4 * 1024, 256, SMEM_BYTES, stream>>>(
        x, Wqkv_bf, b_qkv, Wout_bf, b_out, biasBuf, out);
}
